// EntityExtraction_78030965833780
// MI455X (gfx1250) — matrix operand order repaired and checked by reading, not compiled, not run on hardware
//
#include <hip/hip_runtime.h>
#include <hip/hip_bf16.h>

typedef __attribute__((ext_vector_type(16))) __bf16 v16bf;
typedef __attribute__((ext_vector_type(8)))  float  v8f;
typedef __attribute__((ext_vector_type(4)))  float  f32x4;

#define B_      16
#define T_      512
#define H_      512
#define K_TAGS  21

// ---------------- WMMA fragment helpers (wave32, bf16 16x16x32) -------------

union FragU { v16bf v; f32x4 q[2]; };

// A-matrix 16x32 bf16 from row-major X (ld = ldx). ISA layout:
// lane = g*16+m: elems 0..7 = K[8g..8g+7], 8..15 = K[16+8g..16+8g+7]
__device__ __forceinline__ v16bf load_a(const __bf16* X, int ldx, int row0, int k0, int lane) {
  int g = (lane >> 4) & 1, m = lane & 15;
  const __bf16* p = X + (size_t)(row0 + m) * ldx + (k0 + (g << 3));
  FragU u;
  u.q[0] = *reinterpret_cast<const f32x4*>(p);
  u.q[1] = *reinterpret_cast<const f32x4*>(p + 16);
  return u.v;
}

// B-matrix 32x16 bf16 where B[k][n] = W[n][k], W row-major (N x K, ld = ldw).
// lane = g*16+n: elems 0..7 = K[8g..8g+7], 8..15 = K[16+8g..16+8g+7]
__device__ __forceinline__ v16bf load_b(const __bf16* W, int ldw, int n0, int k0, int lane) {
  int g = (lane >> 4) & 1, n = lane & 15;
  const __bf16* p = W + (size_t)(n0 + n) * ldw + (k0 + (g << 3));
  FragU u;
  u.q[0] = *reinterpret_cast<const f32x4*>(p);
  u.q[1] = *reinterpret_cast<const f32x4*>(p + 16);
  return u.v;
}

__device__ __forceinline__ v8f wmma_bf16(v16bf a, v16bf b, v8f c) {
  return __builtin_amdgcn_wmma_f32_16x16x32_bf16(false, a, false, b, (short)0, c, false, false);
}

__device__ __forceinline__ float sigf(float x) { return 1.f / (1.f + __expf(-x)); }

// ---------------- gfx1250 async global->LDS DMA (ASYNCcnt) -----------------

__device__ __forceinline__ void async_load_b128(unsigned lds_off, const void* gptr) {
  asm volatile("global_load_async_to_lds_b128 %0, %1, off"
               :: "v"(lds_off), "v"((unsigned long long)(size_t)gptr)
               : "memory");
}
__device__ __forceinline__ void wait_async0() {
  asm volatile("s_wait_asynccnt 0" ::: "memory");
}

// ---------------- Generic bf16 WMMA GEMM:  Y = X (MxK) * W^T (W is NxK) + bias
// Register blocking: one wave computes a 32x64 tile (2 A-frags x 4 B-frags ->
// 8 WMMAs per k-step; B-frags reused across both row tiles).

__global__ __launch_bounds__(256) void gemm_bf16(const __bf16* __restrict__ X,
                                                 const __bf16* __restrict__ W,
                                                 const float* __restrict__ bias,
                                                 float* __restrict__ Y,
                                                 int N, int Kdim, int ldy) {
  int lane = threadIdx.x & 31;
  int wid  = blockIdx.x * (blockDim.x >> 5) + (threadIdx.x >> 5);
  int ct4  = N >> 6;
  int rt   = wid / ct4, ct = wid % ct4;
  int row0 = rt << 5, col0 = ct << 6;

  v8f acc[2][4] = {};
  for (int k0 = 0; k0 < Kdim; k0 += 32) {
    v16bf a0 = load_a(X, Kdim, row0,      k0, lane);
    v16bf a1 = load_a(X, Kdim, row0 + 16, k0, lane);
#pragma unroll
    for (int j = 0; j < 4; ++j) {
      v16bf b = load_b(W, Kdim, col0 + 16 * j, k0, lane);
      acc[0][j] = wmma_bf16(a0, b, acc[0][j]);
      acc[1][j] = wmma_bf16(a1, b, acc[1][j]);
    }
  }
  int g = (lane >> 4) & 1, n = lane & 15;
#pragma unroll
  for (int h = 0; h < 2; ++h) {
#pragma unroll
    for (int j = 0; j < 4; ++j) {
      int col = col0 + 16 * j + n;
      float bv = bias ? bias[col] : 0.f;
#pragma unroll
      for (int r = 0; r < 8; ++r) {
        int row = row0 + 16 * h + (g << 3) + r;
        Y[(size_t)row * ldy + col] = acc[h][j][r] + bv;
      }
    }
  }
}

// ---------------- Persistent per-direction LSTM recurrence ------------------
// One workgroup (16 waves / 512 threads) per direction. h (bf16) and c (f32)
// live in LDS. xw[t] slices (16x2048 f32 = 128KB) are double-buffered in LDS
// and prefetched one timestep ahead with async global->LDS DMA; the WMMA GEMM
// (gates += h @ W_hh^T) and the cell nonlinearities hide the DMA latency.
// LDS: 2x128KB gate/xw buffers + 32KB c + 16KB h_bf16 = 304KB (<=320KB WGP).

__global__ __launch_bounds__(512) void lstm_kernel(const float*  __restrict__ xw0,
                                                   const float*  __restrict__ xw1,
                                                   const __bf16* __restrict__ whh0,
                                                   const __bf16* __restrict__ whh1,
                                                   __bf16* __restrict__ out_bf) {
  extern __shared__ char smem[];
  float*  gbuf0 = (float*)smem;                       // 16 x 2048 f32
  float*  gbuf1 = (float*)(smem + 131072);            // 16 x 2048 f32
  float*  cst   = (float*)(smem + 262144);            // 16 x 512  f32
  __bf16* hbf   = (__bf16*)(smem + 262144 + 32768);   // 16 x 512  bf16

  const int dir = blockIdx.x;
  const float*  xw      = dir ? xw1 : xw0;
  const __bf16* whh     = dir ? whh1 : whh0;
  const int     col_off = dir ? H_ : 0;

  int tid  = threadIdx.x;
  int lane = tid & 31;
  int wave = tid >> 5;     // 0..15

  float*   gptr[2] = { gbuf0, gbuf1 };
  unsigned gofs[2] = { (unsigned)(size_t)(void*)gbuf0,
                       (unsigned)(size_t)(void*)gbuf1 };

  for (int p = tid; p < 16 * 512; p += 512) { cst[p] = 0.f; hbf[p] = (__bf16)0.f; }

  // Prefetch xw for the first timestep into buffer 0.
  // 16x2048 f32 slice = 8192 x 16B chunks; 16 waves x 32 lanes x 16 iters.
  {
    int t0 = dir ? (T_ - 1) : 0;
#pragma unroll
    for (int it = 0; it < 16; ++it) {
      int chunk = ((it * 16 + wave) << 5) + lane;       // 0..8191
      int elt = chunk << 2;
      int row = elt >> 11, col = elt & 2047;
      async_load_b128(gofs[0] + ((unsigned)chunk << 4),
                      xw + ((size_t)row * T_ + t0) * 2048 + col);
    }
  }
  wait_async0();
  __syncthreads();

  for (int step = 0; step < T_; ++step) {
    int t = dir ? (T_ - 1 - step) : step;
    int p = step & 1;
    float* gates = gptr[p];

    // Kick off DMA of next timestep's xw slice into the other buffer.
    if (step + 1 < T_) {
      int tn = dir ? (T_ - 2 - step) : (step + 1);
#pragma unroll
      for (int it = 0; it < 16; ++it) {
        int chunk = ((it * 16 + wave) << 5) + lane;
        int elt = chunk << 2;
        int row = elt >> 11, col = elt & 2047;
        async_load_b128(gofs[1 - p] + ((unsigned)chunk << 4),
                        xw + ((size_t)row * T_ + tn) * 2048 + col);
      }
    }

    // gates += h @ W_hh^T (16 x 2048); wave handles 8 N-tiles of 16
    v8f acc[8] = {};
    for (int k0 = 0; k0 < H_; k0 += 32) {
      v16bf a = load_a(hbf, H_, 0, k0, lane);
#pragma unroll
      for (int q = 0; q < 8; ++q) {
        int n0 = ((wave << 3) + q) << 4;
        acc[q] = wmma_bf16(a, load_b(whh, H_, n0, k0, lane), acc[q]);
      }
    }
    int g = (lane >> 4) & 1, n = lane & 15;
#pragma unroll
    for (int q = 0; q < 8; ++q) {
      int col = (((wave << 3) + q) << 4) + n;
#pragma unroll
      for (int r = 0; r < 8; ++r) {
        int row = (g << 3) + r;   // batch index
        gates[row * 2048 + col] += acc[q][r];
      }
    }
    __syncthreads();

    // elementwise LSTM cell
    for (int p2 = tid; p2 < 16 * 512; p2 += 512) {
      int b = p2 >> 9, j = p2 & 511;
      float gi = gates[b * 2048 + j];
      float gf = gates[b * 2048 + j + 512];
      float gg = gates[b * 2048 + j + 1024];
      float go = gates[b * 2048 + j + 1536];
      float c  = sigf(gf) * cst[p2] + sigf(gi) * tanhf(gg);
      cst[p2] = c;
      float h = sigf(go) * tanhf(c);
      hbf[p2] = (__bf16)h;
      out_bf[((size_t)b * T_ + t) * 1024 + col_off + j] = (__bf16)h;
    }
    // Own async DMAs done; barrier publishes every wave's chunk to all waves.
    wait_async0();
    __syncthreads();
  }
}

// ---------------- Small kernels --------------------------------------------

__global__ void cvt_kernel(const float* __restrict__ s, __bf16* __restrict__ d, int n) {
  for (int i = blockIdx.x * blockDim.x + threadIdx.x; i < n; i += gridDim.x * blockDim.x)
    d[i] = (__bf16)s[i];
}

__global__ void bias2_kernel(const float* __restrict__ a, const float* __restrict__ b,
                             float* __restrict__ o, int n) {
  int i = blockIdx.x * blockDim.x + threadIdx.x;
  if (i < n) o[i] = a[i] + b[i];
}

__global__ void embed_kernel(const int* __restrict__ xw, const int* __restrict__ xp,
                             const float* __restrict__ wemb, const float* __restrict__ pemb,
                             __bf16* __restrict__ feat) {
  int id = blockIdx.x * blockDim.x + threadIdx.x;
  if (id >= B_ * T_ * 512) return;
  int bt = id >> 9, c = id & 511;
  float v = (c < 256) ? wemb[(size_t)xw[bt] * 256 + c]
                      : pemb[(size_t)xp[bt] * 256 + (c - 256)];
  feat[id] = (__bf16)v;
}

// ner_out = tmp (8192x128) @ W2^T (21x128) + b2
__global__ void proj2_kernel(const float* __restrict__ tmp, const float* __restrict__ W2,
                             const float* __restrict__ b2, float* __restrict__ out, int total) {
  int id = blockIdx.x * blockDim.x + threadIdx.x;
  if (id >= total) return;
  int row = id / K_TAGS, c = id % K_TAGS;
  float s = b2[c];
  const float* tp = tmp + (size_t)row * 128;
  const float* wp = W2 + c * 128;
#pragma unroll 8
  for (int k = 0; k < 128; ++k) s += tp[k] * wp[k];
  out[id] = s;
}

// ---------------- CRF Viterbi decode (one wave32 per batch row) -------------

__global__ void viterbi_kernel(const float* __restrict__ em,
                               const float* __restrict__ start, const float* __restrict__ endv,
                               const float* __restrict__ trans,
                               int* __restrict__ hist, float* __restrict__ dec_out) {
  int b = blockIdx.x, lane = threadIdx.x;
  bool act = lane < K_TAGS;
  const float NEG = -1e30f;
  float score = act ? start[lane] + em[((size_t)b * T_) * K_TAGS + lane] : NEG;
  for (int t = 1; t < T_; ++t) {
    float best = NEG; int barg = 0;
    for (int i = 0; i < K_TAGS; ++i) {
      float si   = __shfl(score, i, 32);
      float cand = si + (act ? trans[i * K_TAGS + lane] : 0.f);
      if (cand > best) { best = cand; barg = i; }
    }
    if (act) hist[((size_t)b * (T_ - 1) + (t - 1)) * K_TAGS + lane] = barg;
    score = act ? best + em[((size_t)b * T_ + t) * K_TAGS + lane] : NEG;
  }
  float fin = act ? score + endv[lane] : NEG;
  int best_tag = 0; float bv = NEG;
  for (int i = 0; i < K_TAGS; ++i) {
    float v = __shfl(fin, i, 32);
    if (v > bv) { bv = v; best_tag = i; }
  }
  if (lane == 0) {
    int tag = best_tag;
    dec_out[(size_t)b * T_ + (T_ - 1)] = (float)tag;
    for (int t = T_ - 2; t >= 0; --t) {
      tag = hist[((size_t)b * (T_ - 1) + t) * K_TAGS + tag];
      dec_out[(size_t)b * T_ + t] = (float)tag;
    }
  }
}

// ---------------- CRF loss (16 waves, one batch row each) -------------------

__global__ __launch_bounds__(512) void crf_kernel(const float* __restrict__ em,
                                                  const int* __restrict__ tags,
                                                  const float* __restrict__ start,
                                                  const float* __restrict__ endv,
                                                  const float* __restrict__ trans,
                                                  float* __restrict__ loss_out) {
  __shared__ float llh_s[16];
  int tid = threadIdx.x, lane = tid & 31, b = tid >> 5;
  bool act = lane < K_TAGS;
  const float NEG = -1e30f;
  float score = act ? start[lane] + em[((size_t)b * T_) * K_TAGS + lane] : NEG;
  for (int t = 1; t < T_; ++t) {
    float m = NEG;
    for (int i = 0; i < K_TAGS; ++i) {
      float si = __shfl(score, i, 32);
      m = fmaxf(m, si + (act ? trans[i * K_TAGS + lane] : NEG));
    }
    float s = 0.f;
    for (int i = 0; i < K_TAGS; ++i) {
      float si = __shfl(score, i, 32);
      s += __expf(si + (act ? trans[i * K_TAGS + lane] : NEG) - m);
    }
    score = act ? m + __logf(s) + em[((size_t)b * T_ + t) * K_TAGS + lane] : NEG;
  }
  float fin = act ? score + endv[lane] : NEG;
  float m2 = NEG;
  for (int i = 0; i < K_TAGS; ++i) m2 = fmaxf(m2, __shfl(fin, i, 32));
  float s2 = 0.f;
  for (int i = 0; i < K_TAGS; ++i) s2 += __expf(__shfl(fin, i, 32) - m2);
  float denom = m2 + __logf(s2);
  if (lane == 0) {
    const int* tg = tags + (size_t)b * T_;
    float num = start[tg[0]] + em[((size_t)b * T_) * K_TAGS + tg[0]];
    for (int t = 1; t < T_; ++t)
      num += trans[tg[t - 1] * K_TAGS + tg[t]] + em[((size_t)b * T_ + t) * K_TAGS + tg[t]];
    num += endv[tg[T_ - 1]];
    llh_s[b] = num - denom;
  }
  __syncthreads();
  if (tid == 0) {
    float sum = 0.f;
    for (int i = 0; i < 16; ++i) sum += llh_s[i];
    loss_out[0] = -(sum / (float)(B_ * T_));
  }
}

// ---------------- Host orchestration ---------------------------------------

extern "C" void kernel_launch(void* const* d_in, const int* in_sizes, int n_in,
                              void* d_out, int out_size, void* d_ws, size_t ws_size,
                              hipStream_t stream) {
  const int*   x_word = (const int*)d_in[0];
  const int*   x_pos  = (const int*)d_in[1];
  /* d_in[2] = mask (all ones, unused) */
  const int*   y_word = (const int*)d_in[3];
  const float* wemb   = (const float*)d_in[4];
  const float* pemb   = (const float*)d_in[5];
  const float* wih1f = (const float*)d_in[6],  *whh1f = (const float*)d_in[7];
  const float* bih1f = (const float*)d_in[8],  *bhh1f = (const float*)d_in[9];
  const float* wih1b = (const float*)d_in[10], *whh1b = (const float*)d_in[11];
  const float* bih1b = (const float*)d_in[12], *bhh1b = (const float*)d_in[13];
  const float* wih2f = (const float*)d_in[14], *whh2f = (const float*)d_in[15];
  const float* bih2f = (const float*)d_in[16], *bhh2f = (const float*)d_in[17];
  const float* wih2b = (const float*)d_in[18], *whh2b = (const float*)d_in[19];
  const float* bih2b = (const float*)d_in[20], *bhh2b = (const float*)d_in[21];
  const float* W1 = (const float*)d_in[22], *b1 = (const float*)d_in[23];
  const float* W2 = (const float*)d_in[24], *b2 = (const float*)d_in[25];
  const float* crf_start = (const float*)d_in[26];
  const float* crf_end   = (const float*)d_in[27];
  const float* crf_trans = (const float*)d_in[28];

  char* ws = (char*)d_ws;
  size_t off = 0;
  auto alloc = [&](size_t bytes) -> void* {
    void* p = (void*)(ws + off);
    off += (bytes + 255) & ~(size_t)255;
    return p;
  };

  __bf16* feat_bf  = (__bf16*)alloc((size_t)8192 * 512 * 2);
  __bf16* wih1f_b  = (__bf16*)alloc((size_t)2048 * 512 * 2);
  __bf16* wih1b_b  = (__bf16*)alloc((size_t)2048 * 512 * 2);
  __bf16* whh1f_b  = (__bf16*)alloc((size_t)2048 * 512 * 2);
  __bf16* whh1b_b  = (__bf16*)alloc((size_t)2048 * 512 * 2);
  __bf16* wih2f_b  = (__bf16*)alloc((size_t)2048 * 1024 * 2);
  __bf16* wih2b_b  = (__bf16*)alloc((size_t)2048 * 1024 * 2);
  __bf16* whh2f_b  = (__bf16*)alloc((size_t)2048 * 512 * 2);
  __bf16* whh2b_b  = (__bf16*)alloc((size_t)2048 * 512 * 2);
  __bf16* W1_b     = (__bf16*)alloc((size_t)128 * 1024 * 2);
  float*  bias1f   = (float*)alloc((size_t)2048 * 4);
  float*  bias1b   = (float*)alloc((size_t)2048 * 4);
  float*  bias2f   = (float*)alloc((size_t)2048 * 4);
  float*  bias2b   = (float*)alloc((size_t)2048 * 4);
  float*  xw_f     = (float*)alloc((size_t)8192 * 2048 * 4);
  float*  xw_b     = (float*)alloc((size_t)8192 * 2048 * 4);
  __bf16* h1_bf    = (__bf16*)alloc((size_t)8192 * 1024 * 2);
  __bf16* h2_bf    = (__bf16*)alloc((size_t)8192 * 1024 * 2);
  float*  tmp      = (float*)alloc((size_t)8192 * 128 * 4);
  int*    hist     = (int*)alloc((size_t)16 * (T_ - 1) * K_TAGS * 4);

  float* out_f  = (float*)d_out;
  float* ner    = out_f;                 // 16*512*21
  float* dec    = out_f + 172032;        // 16*512
  float* loss   = out_f + 172032 + 8192; // 1

  const int SMEM = 2 * 131072 + 32768 + 16384;   // 311296 B <= 320KB WGP LDS
  (void)hipFuncSetAttribute((const void*)lstm_kernel,
                            hipFuncAttributeMaxDynamicSharedMemorySize, SMEM);

  // --- convert weights / fold biases
  auto cvt = [&](const float* s, __bf16* d, int n) {
    cvt_kernel<<<(n + 255) / 256, 256, 0, stream>>>(s, d, n);
  };
  cvt(wih1f, wih1f_b, 2048 * 512);  cvt(wih1b, wih1b_b, 2048 * 512);
  cvt(whh1f, whh1f_b, 2048 * 512);  cvt(whh1b, whh1b_b, 2048 * 512);
  cvt(wih2f, wih2f_b, 2048 * 1024); cvt(wih2b, wih2b_b, 2048 * 1024);
  cvt(whh2f, whh2f_b, 2048 * 512);  cvt(whh2b, whh2b_b, 2048 * 512);
  cvt(W1, W1_b, 128 * 1024);
  bias2_kernel<<<8, 256, 0, stream>>>(bih1f, bhh1f, bias1f, 2048);
  bias2_kernel<<<8, 256, 0, stream>>>(bih1b, bhh1b, bias1b, 2048);
  bias2_kernel<<<8, 256, 0, stream>>>(bih2f, bhh2f, bias2f, 2048);
  bias2_kernel<<<8, 256, 0, stream>>>(bih2b, bhh2b, bias2b, 2048);

  // --- embeddings -> bf16 features (8192 x 512)
  embed_kernel<<<(8192 * 512 + 255) / 256, 256, 0, stream>>>(x_word, x_pos, wemb, pemb, feat_bf);

  // --- layer 1: xw = feat @ W_ih^T + (b_ih+b_hh)   [8192x512 * 512x2048]
  // waves = (8192/32)*(2048/64) = 8192 -> 1024 blocks of 8 waves
  gemm_bf16<<<1024, 256, 0, stream>>>(feat_bf, wih1f_b, bias1f, xw_f, 2048, 512, 2048);
  gemm_bf16<<<1024, 256, 0, stream>>>(feat_bf, wih1b_b, bias1b, xw_b, 2048, 512, 2048);
  lstm_kernel<<<2, 512, SMEM, stream>>>(xw_f, xw_b, whh1f_b, whh1b_b, h1_bf);

  // --- layer 2: xw = h1 @ W_ih^T + bias   [8192x1024 * 1024x2048]
  gemm_bf16<<<1024, 256, 0, stream>>>(h1_bf, wih2f_b, bias2f, xw_f, 2048, 1024, 2048);
  gemm_bf16<<<1024, 256, 0, stream>>>(h1_bf, wih2b_b, bias2b, xw_b, 2048, 1024, 2048);
  lstm_kernel<<<2, 512, SMEM, stream>>>(xw_f, xw_b, whh2f_b, whh2b_b, h2_bf);

  // --- head: tmp = h2 @ W1^T + b1  [8192x1024 * 1024x128]; waves=256*2 -> 64 blocks
  gemm_bf16<<<64, 256, 0, stream>>>(h2_bf, W1_b, b1, tmp, 128, 1024, 128);
  proj2_kernel<<<(172032 + 255) / 256, 256, 0, stream>>>(tmp, W2, b2, ner, 172032);

  // --- CRF decode + loss
  viterbi_kernel<<<16, 32, 0, stream>>>(ner, crf_start, crf_end, crf_trans, hist, dec);
  crf_kernel<<<1, 512, 0, stream>>>(ner, y_word, crf_start, crf_end, crf_trans, loss);
}
